// PointNet2Regressor_15324443312191
// MI455X (gfx1250) — matrix operand order repaired and checked by reading, not compiled, not run on hardware
//
#include <hip/hip_runtime.h>
#include <hip/hip_fp16.h>
#include <hip/hip_bf16.h>

// ---------------------------------------------------------------------------
// PointNet++ regressor for MI455X (gfx1250, wave32, WMMA).
// All matmuls go through v_wmma_f32_16x16x32_f16 (f16 inputs, f32 accum).
// Weights are pre-swizzled into per-lane WMMA B-fragment order so the GEMM
// inner loop is two b128 loads for A, two b128 loads for B, one WMMA.
// Workspace requirement: ~412 MB (see layout in kernel_launch).
// ---------------------------------------------------------------------------

typedef __attribute__((ext_vector_type(16))) _Float16 v16h;
typedef __attribute__((ext_vector_type(8)))  _Float16 v8h;
typedef __attribute__((ext_vector_type(8)))  float    v8f;

#define PN_B     32
#define PN_N     2048
#define PN_S1    512
#define PN_NS1   32
#define PN_S2    128
#define PN_NS2   64
#define PN_OUTV  1500
#define PN_EPS   1e-5f

// ---------------- transpose (B,3,N) -> (B,N,3) ----------------
__global__ void pn_transpose_k(const float* __restrict__ xyz, float* __restrict__ pts) {
  int t = blockIdx.x * blockDim.x + threadIdx.x;          // over B*N
  if (t >= PN_B * PN_N) return;
  int b = t / PN_N, n = t % PN_N;
  const float* src = xyz + (size_t)b * 3 * PN_N + n;
  float* dst = pts + (size_t)t * 3;
  dst[0] = src[0];
  dst[1] = src[PN_N];
  dst[2] = src[2 * PN_N];
}

// ---------------- farthest point sampling (one block per batch) ------------
__global__ __launch_bounds__(256)
void pn_fps_k(const float* __restrict__ pts, int N, int npoint,
              int* __restrict__ fidx, float* __restrict__ nxyz) {
  __shared__ float dist[PN_N];
  __shared__ float rv[256];
  __shared__ int   ri[256];
  __shared__ int   sfar;
  const int b = blockIdx.x;
  const float* P = pts + (size_t)b * N * 3;
  for (int i = threadIdx.x; i < N; i += blockDim.x) dist[i] = 1e10f;
  if (threadIdx.x == 0) sfar = 0;
  __syncthreads();
  for (int it = 0; it < npoint; ++it) {
    const int far = sfar;
    const float cx = P[far * 3 + 0], cy = P[far * 3 + 1], cz = P[far * 3 + 2];
    if (threadIdx.x == 0) {
      fidx[b * npoint + it] = far;
      float* o = nxyz + (size_t)(b * npoint + it) * 3;
      o[0] = cx; o[1] = cy; o[2] = cz;
    }
    float best = -1.0f; int bi = 0;
    for (int i = threadIdx.x; i < N; i += blockDim.x) {
      float dx = P[i * 3 + 0] - cx;
      float dy = P[i * 3 + 1] - cy;
      float dz = P[i * 3 + 2] - cz;
      float d  = dx * dx + dy * dy + dz * dz;
      float nd = fminf(dist[i], d);
      dist[i] = nd;
      if (nd > best) { best = nd; bi = i; }   // strict > keeps lowest index
    }
    rv[threadIdx.x] = best; ri[threadIdx.x] = bi;
    __syncthreads();
    for (int s = 128; s > 0; s >>= 1) {
      if (threadIdx.x < s) {
        float v2 = rv[threadIdx.x + s]; int i2 = ri[threadIdx.x + s];
        if (v2 > rv[threadIdx.x] || (v2 == rv[threadIdx.x] && i2 < ri[threadIdx.x])) {
          rv[threadIdx.x] = v2; ri[threadIdx.x] = i2;
        }
      }
      __syncthreads();
    }
    if (threadIdx.x == 0) sfar = ri[0];
    __syncthreads();
  }
}

// ---------------- ball query: first `ns` in-range indices, pad with first --
__global__ void pn_ballq_k(const float* __restrict__ xyz, const float* __restrict__ nx,
                           int N, int S, int ns, float r2, int* __restrict__ gidx,
                           int total) {
  int t = blockIdx.x * blockDim.x + threadIdx.x;          // over B*S
  if (t >= total) return;
  int b = t / S;
  const float* P = xyz + (size_t)b * N * 3;
  const float cx = nx[t * 3 + 0], cy = nx[t * 3 + 1], cz = nx[t * 3 + 2];
  int* g = gidx + (size_t)t * ns;
  int cnt = 0, first = 0;
  for (int j = 0; j < N && cnt < ns; ++j) {
    float dx = P[j * 3 + 0] - cx;
    float dy = P[j * 3 + 1] - cy;
    float dz = P[j * 3 + 2] - cz;
    if (dx * dx + dy * dy + dz * dz <= r2) {
      if (cnt == 0) first = j;
      g[cnt++] = j;
    }
  }
  for (; cnt < ns; ++cnt) g[cnt] = first;
}

// ---------------- grouping kernels: build padded f16 activation rows -------
__global__ void pn_group1_k(const float* __restrict__ pts, const float* __restrict__ nxyz,
                            const int* __restrict__ gidx, _Float16* __restrict__ X) {
  int t = blockIdx.x * blockDim.x + threadIdx.x;          // over B*S1*NS1
  if (t >= PN_B * PN_S1 * PN_NS1) return;
  int q = t / PN_NS1;                                     // (b*S1+s)
  int b = q / PN_S1;
  int g = gidx[t];
  const float* p = pts + ((size_t)b * PN_N + g) * 3;
  const float* c = nxyz + (size_t)q * 3;
  _Float16* x = X + (size_t)t * 32;                       // Kpad = 32
  x[0] = (_Float16)(p[0] - c[0]);
  x[1] = (_Float16)(p[1] - c[1]);
  x[2] = (_Float16)(p[2] - c[2]);
  for (int i = 3; i < 32; ++i) x[i] = (_Float16)0.0f;
}

__global__ void pn_group2_k(const float* __restrict__ l1x, const float* __restrict__ nxyz2,
                            const float* __restrict__ l1p, const int* __restrict__ gidx,
                            _Float16* __restrict__ X) {
  int t = blockIdx.x * blockDim.x + threadIdx.x;          // over B*S2*NS2
  if (t >= PN_B * PN_S2 * PN_NS2) return;
  int q = t / PN_NS2;                                     // (b*S2+s)
  int b = q / PN_S2;
  int g = gidx[t];
  const float* p = l1x + ((size_t)b * PN_S1 + g) * 3;
  const float* c = nxyz2 + (size_t)q * 3;
  const float* f = l1p + ((size_t)b * PN_S1 + g) * 128;
  _Float16* x = X + (size_t)t * 160;                      // Kpad = 160 (131 real)
  x[0] = (_Float16)(p[0] - c[0]);
  x[1] = (_Float16)(p[1] - c[1]);
  x[2] = (_Float16)(p[2] - c[2]);
  for (int i = 0; i < 128; ++i) x[3 + i] = (_Float16)f[i];
  for (int i = 131; i < 160; ++i) x[i] = (_Float16)0.0f;
}

__global__ void pn_group3_k(const float* __restrict__ l2x, const float* __restrict__ l2p,
                            _Float16* __restrict__ X) {
  int t = blockIdx.x * blockDim.x + threadIdx.x;          // over B*S2
  if (t >= PN_B * PN_S2) return;
  const float* p = l2x + (size_t)t * 3;
  const float* f = l2p + (size_t)t * 256;
  _Float16* x = X + (size_t)t * 288;                      // Kpad = 288 (259 real)
  x[0] = (_Float16)p[0];
  x[1] = (_Float16)p[1];
  x[2] = (_Float16)p[2];
  for (int i = 0; i < 256; ++i) x[3 + i] = (_Float16)f[i];
  for (int i = 259; i < 288; ++i) x[i] = (_Float16)0.0f;
}

// ---------------- weight pack: f32 (K,N) -> per-lane WMMA B fragments ------
// Fragment layout: lane L of tile (nt,kt) holds N = nt*16 + (L&15),
// K = kt*32 + (L<16?0:16) + h for h=0..15 (contiguous!).
__global__ void pn_packw_k(const float* __restrict__ W, _Float16* __restrict__ Wp,
                           int Kreal, int Nreal, int Kpad) {
  const int Kt = Kpad >> 5;
  const int Nt = (Nreal + 15) >> 4;
  int t = blockIdx.x * blockDim.x + threadIdx.x;          // over Nt*Kt*32
  if (t >= Nt * Kt * 32) return;
  int lane = t & 31;
  int kt = (t >> 5) % Kt;
  int n = ((t >> 5) / Kt) * 16 + (lane & 15);
  int koff = kt * 32 + ((lane < 16) ? 0 : 16);
  _Float16* dst = Wp + (size_t)t * 16;
  for (int h = 0; h < 16; ++h) {
    int k = koff + h;
    float v = (k < Kreal && n < Nreal) ? W[(size_t)k * Nreal + n] : 0.0f;
    dst[h] = (_Float16)v;
  }
}

// ---------------- WMMA GEMM: Y[rows,N] = X[rows,Kpad] * W + bias -----------
__global__ __launch_bounds__(128)
void pn_wmma_gemm_k(const _Float16* __restrict__ X, const _Float16* __restrict__ Wp,
                    const float* __restrict__ bias, float* __restrict__ Y,
                    int rows, int Kpad, int Nreal) {
  const int lane = threadIdx.x & 31;
  const int wave = threadIdx.x >> 5;
  const int mtiles = rows >> 4;
  const int ntiles = (Nreal + 15) >> 4;
  long tile = (long)blockIdx.x * 4 + wave;
  if (tile >= (long)mtiles * ntiles) return;
  const int mt = (int)(tile % mtiles);
  const int nt = (int)(tile / mtiles);
  const int Kt = Kpad >> 5;
  const int row = (mt << 4) + (lane & 15);
  const int abase = (lane < 16) ? 0 : 16;                 // A half-wave K offset
  const _Float16* ap = X + (size_t)row * Kpad + abase;
  const _Float16* bp = Wp + (((size_t)nt * Kt) * 32 + lane) * 16;
  v8f acc = {};
  for (int kt = 0; kt < Kt; ++kt) {
    v8h a0 = *(const v8h*)(ap);           // K = k0+abase .. +7
    v8h a1 = *(const v8h*)(ap + 8);       // K = k0+abase+8 .. +15
    v8h b0 = *(const v8h*)(bp);
    v8h b1 = *(const v8h*)(bp + 8);
    __builtin_prefetch(bp + 512, 0, 1);   // next kt weight fragment (global_prefetch_b8)
    v16h A  = __builtin_shufflevector(a0, a1, 0,1,2,3,4,5,6,7,8,9,10,11,12,13,14,15);
    v16h Bm = __builtin_shufflevector(b0, b1, 0,1,2,3,4,5,6,7,8,9,10,11,12,13,14,15);
    acc = __builtin_amdgcn_wmma_f32_16x16x32_f16(false, A, false, Bm,
                                                 (short)0, acc, false, false);
    ap += 32;
    bp += 512;                             // 32 lanes * 16 halfs
  }
  const int n = (nt << 4) + (lane & 15);
  if (n < Nreal) {
    const float bv = bias[n];
    const int m0 = (mt << 4) + ((lane < 16) ? 0 : 8);     // C layout: VGPR r -> M = m0+r
#pragma unroll
    for (int r = 0; r < 8; ++r)
      Y[(size_t)(m0 + r) * Nreal + n] = acc[r] + bv;
  }
}

// ---------------- BN stats: per-column sum / sumsq (f32 atomics) -----------
__global__ void pn_zero_k(float* __restrict__ p, int n) {
  int t = blockIdx.x * blockDim.x + threadIdx.x;
  if (t < n) p[t] = 0.0f;
}

__global__ void pn_colstats_k(const float* __restrict__ Y, float* __restrict__ stats,
                              int rows, int N) {
  int c = blockIdx.x * 64 + threadIdx.x;
  if (c >= N) return;
  float s = 0.0f, ss = 0.0f;
  for (int r = blockIdx.y; r < rows; r += gridDim.y) {
    float v = Y[(size_t)r * N + c];
    s += v; ss += v * v;
  }
  atomicAdd(&stats[c], s);
  atomicAdd(&stats[N + c], ss);
}

// ---------------- BN + ReLU + convert to padded f16 ------------------------
__global__ void pn_bnrelu_k(const float* __restrict__ Y, const float* __restrict__ stats,
                            const float* __restrict__ gamma, const float* __restrict__ beta,
                            _Float16* __restrict__ X, int rows, int N, int Kpadn) {
  size_t t = (size_t)blockIdx.x * blockDim.x + threadIdx.x;
  size_t total = (size_t)rows * Kpadn;
  if (t >= total) return;
  int c = (int)(t % Kpadn);
  size_t r = t / Kpadn;
  float val = 0.0f;
  if (c < N) {
    float inv = 1.0f / (float)rows;
    float m = stats[c] * inv;
    float v = stats[N + c] * inv - m * m;
    float y = Y[r * N + c];
    val = (y - m) * rsqrtf(v + PN_EPS) * gamma[c] + beta[c];
    val = fmaxf(val, 0.0f);
  }
  X[t] = (_Float16)val;
}

// ---------------- max-pool over samples ------------------------------------
__global__ void pn_pool_k(const _Float16* __restrict__ X, float* __restrict__ out,
                          int groups, int ns, int C, int Cpad) {
  int t = blockIdx.x * blockDim.x + threadIdx.x;          // over groups*C
  if (t >= groups * C) return;
  int c = t % C, g = t / C;
  float m = -3.4e38f;
  for (int k = 0; k < ns; ++k)
    m = fmaxf(m, (float)X[((size_t)g * ns + k) * Cpad + c]);
  out[t] = m;
}

__global__ void pn_cvt_k(const float* __restrict__ s, _Float16* __restrict__ d, int n) {
  int t = blockIdx.x * blockDim.x + threadIdx.x;
  if (t < n) d[t] = (_Float16)s[t];
}

// ---------------- output head: trans + normalized tanh ---------------------
__global__ void pn_head_k(const float* __restrict__ yt, const float* __restrict__ yn,
                          float* __restrict__ out, int total) {
  int t = blockIdx.x * blockDim.x + threadIdx.x;          // over B*OUTV
  if (t >= total) return;
  int b = t / PN_OUTV, i = t % PN_OUTV;
  const float* pt = yt + (size_t)b * (PN_OUTV * 3) + i * 3;
  const float* pnr = yn + (size_t)b * (PN_OUTV * 3) + i * 3;
  float n0 = tanhf(pnr[0]), n1 = tanhf(pnr[1]), n2 = tanhf(pnr[2]);
  float len = fmaxf(sqrtf(n0 * n0 + n1 * n1 + n2 * n2), 1e-12f);
  float* o = out + (size_t)t * 6;
  o[0] = pt[0]; o[1] = pt[1]; o[2] = pt[2];
  o[3] = n0 / len; o[4] = n1 / len; o[5] = n2 / len;
}

// ---------------------------------------------------------------------------
extern "C" void kernel_launch(void* const* d_in, const int* in_sizes, int n_in,
                              void* d_out, int out_size, void* d_ws, size_t ws_size,
                              hipStream_t stream) {
  (void)in_sizes; (void)n_in; (void)out_size; (void)ws_size;
  const float* xyz = (const float*)d_in[0];
  struct Layer { const float *w, *b, *g, *be; int din, dout; };
  Layer L[9];
  static const int dims[9][2] = {{3,64},{64,64},{64,128},
                                 {131,128},{128,128},{128,256},
                                 {259,256},{256,512},{512,1024}};
  for (int i = 0; i < 9; ++i) {
    L[i].w  = (const float*)d_in[1 + i * 4 + 0];
    L[i].b  = (const float*)d_in[1 + i * 4 + 1];
    L[i].g  = (const float*)d_in[1 + i * 4 + 2];
    L[i].be = (const float*)d_in[1 + i * 4 + 3];
    L[i].din = dims[i][0]; L[i].dout = dims[i][1];
  }
  const float* fc1_w = (const float*)d_in[37]; const float* fc1_b = (const float*)d_in[38];
  const float* bn1_g = (const float*)d_in[39]; const float* bn1_b = (const float*)d_in[40];
  const float* fc2_w = (const float*)d_in[41]; const float* fc2_b = (const float*)d_in[42];
  const float* bn2_g = (const float*)d_in[43]; const float* bn2_b = (const float*)d_in[44];
  const float* fc3_w = (const float*)d_in[45]; const float* fc3_b = (const float*)d_in[46];
  const float* fcn_w = (const float*)d_in[47]; const float* fcn_b = (const float*)d_in[48];
  float* out = (float*)d_out;

  // ---- workspace layout (total ~412 MB) ----
  char* cur = (char*)d_ws;
  auto alloc = [&](size_t bytes) -> void* {
    void* p = (void*)cur; cur += (bytes + 255) & ~(size_t)255; return p;
  };
  float*     pts   = (float*)alloc((size_t)PN_B * PN_N * 3 * 4);
  int*       fidx1 = (int*)alloc((size_t)PN_B * PN_S1 * 4);
  float*     nxyz1 = (float*)alloc((size_t)PN_B * PN_S1 * 3 * 4);
  int*       gidx1 = (int*)alloc((size_t)PN_B * PN_S1 * PN_NS1 * 4);
  int*       fidx2 = (int*)alloc((size_t)PN_B * PN_S2 * 4);
  float*     nxyz2 = (float*)alloc((size_t)PN_B * PN_S2 * 3 * 4);
  int*       gidx2 = (int*)alloc((size_t)PN_B * PN_S2 * PN_NS2 * 4);
  float*     l1p   = (float*)alloc((size_t)PN_B * PN_S1 * 128 * 4);
  float*     l2p   = (float*)alloc((size_t)PN_B * PN_S2 * 256 * 4);
  float*     gf    = (float*)alloc((size_t)PN_B * 1024 * 4);
  float*     stats = (float*)alloc((size_t)2 * 1024 * 4);
  float*     yt    = (float*)alloc((size_t)PN_B * 4500 * 4);
  float*     yn    = (float*)alloc((size_t)PN_B * 4500 * 4);
  _Float16*  Wp    = (_Float16*)alloc((size_t)10 * 1024 * 1024);
  _Float16*  X     = (_Float16*)alloc((size_t)524288 * 128 * 2);
  float*     Y     = (float*)alloc((size_t)524288 * 128 * 4);

  auto ceil32 = [](int x) { return (x + 31) & ~31; };
  auto run_gemm = [&](const float* w, const float* bias, int Kreal, int Kpad,
                      int Nreal, int rows, float* Yd) {
    int Kt = Kpad >> 5, Nt = (Nreal + 15) >> 4;
    int pth = Nt * Kt * 32;
    pn_packw_k<<<(pth + 255) / 256, 256, 0, stream>>>(w, Wp, Kreal, Nreal, Kpad);
    long tiles = (long)(rows >> 4) * Nt;
    pn_wmma_gemm_k<<<(unsigned)((tiles + 3) / 4), 128, 0, stream>>>(
        X, Wp, bias, Yd, rows, Kpad, Nreal);
  };
  auto run_bn = [&](const float* g, const float* be, int Nreal, int rows) {
    int Kn = ceil32(Nreal);
    pn_zero_k<<<(2 * Nreal + 255) / 256, 256, 0, stream>>>(stats, 2 * Nreal);
    dim3 gs((Nreal + 63) / 64, (unsigned)(rows < 256 ? rows : 256));
    pn_colstats_k<<<gs, 64, 0, stream>>>(Y, stats, rows, Nreal);
    size_t tot = (size_t)rows * Kn;
    pn_bnrelu_k<<<(unsigned)((tot + 255) / 256), 256, 0, stream>>>(
        Y, stats, g, be, X, rows, Nreal, Kn);
  };
  auto run_layer = [&](const Layer& l, int Kpad, int rows) {
    run_gemm(l.w, l.b, l.din, Kpad, l.dout, rows, Y);
    run_bn(l.g, l.be, l.dout, rows);
  };

  // ---- stage 0: transpose ----
  pn_transpose_k<<<(PN_B * PN_N + 255) / 256, 256, 0, stream>>>(xyz, pts);

  // ---- SA1: FPS 2048->512, ball r=0.2 ns=32, MLP 3->64->64->128 ----
  pn_fps_k<<<PN_B, 256, 0, stream>>>(pts, PN_N, PN_S1, fidx1, nxyz1);
  pn_ballq_k<<<(PN_B * PN_S1 + 255) / 256, 256, 0, stream>>>(
      pts, nxyz1, PN_N, PN_S1, PN_NS1, 0.04f, gidx1, PN_B * PN_S1);
  {
    int rows = PN_B * PN_S1 * PN_NS1;                     // 524288
    pn_group1_k<<<(rows + 255) / 256, 256, 0, stream>>>(pts, nxyz1, gidx1, X);
    run_layer(L[0], 32, rows);
    run_layer(L[1], 64, rows);
    run_layer(L[2], 64, rows);
    int groups = PN_B * PN_S1;
    pn_pool_k<<<(groups * 128 + 255) / 256, 256, 0, stream>>>(X, l1p, groups, PN_NS1, 128, 128);
  }

  // ---- SA2: FPS 512->128, ball r=0.4 ns=64, MLP 131->128->128->256 ----
  pn_fps_k<<<PN_B, 256, 0, stream>>>(nxyz1, PN_S1, PN_S2, fidx2, nxyz2);
  pn_ballq_k<<<(PN_B * PN_S2 + 255) / 256, 256, 0, stream>>>(
      nxyz1, nxyz2, PN_S1, PN_S2, PN_NS2, 0.16f, gidx2, PN_B * PN_S2);
  {
    int rows = PN_B * PN_S2 * PN_NS2;                     // 262144
    pn_group2_k<<<(rows + 255) / 256, 256, 0, stream>>>(nxyz1, nxyz2, l1p, gidx2, X);
    run_layer(L[3], 160, rows);
    run_layer(L[4], 128, rows);
    run_layer(L[5], 128, rows);
    int groups = PN_B * PN_S2;
    pn_pool_k<<<(groups * 256 + 255) / 256, 256, 0, stream>>>(X, l2p, groups, PN_NS2, 256, 256);
  }

  // ---- SA3: group_all, MLP 259->256->512->1024, max over 128 pts ----
  {
    int rows = PN_B * PN_S2;                              // 4096
    pn_group3_k<<<(rows + 255) / 256, 256, 0, stream>>>(nxyz2, l2p, X);
    run_layer(L[6], 288, rows);
    run_layer(L[7], 256, rows);
    run_layer(L[8], 512, rows);
    pn_pool_k<<<(PN_B * 1024 + 255) / 256, 256, 0, stream>>>(X, gf, PN_B, PN_S2, 1024, 1024);
  }

  // ---- FC head ----
  pn_cvt_k<<<(PN_B * 1024 + 255) / 256, 256, 0, stream>>>(gf, X, PN_B * 1024);
  run_gemm(fc1_w, fc1_b, 1024, 1024, 1024, PN_B, Y);
  run_bn(bn1_g, bn1_b, 1024, PN_B);
  run_gemm(fc2_w, fc2_b, 1024, 1024, 1024, PN_B, Y);
  run_bn(bn2_g, bn2_b, 1024, PN_B);
  run_gemm(fc3_w, fc3_b, 1024, 1024, 4500, PN_B, yt);
  run_gemm(fcn_w, fcn_b, 1024, 1024, 4500, PN_B, yn);

  pn_head_k<<<(PN_B * PN_OUTV + 255) / 256, 256, 0, stream>>>(yt, yn, out, PN_B * PN_OUTV);
}
